// SampleModel_27779848471082
// MI455X (gfx1250) — matrix operand order repaired and checked by reading, not compiled, not run on hardware
//
#include <hip/hip_runtime.h>
#include <hip/hip_bf16.h>

typedef __attribute__((ext_vector_type(16))) _Float16 v16h;
typedef __attribute__((ext_vector_type(8)))  _Float16 v8h;
typedef __attribute__((ext_vector_type(8)))  float    v8f;
typedef _Float16 half_t;

static inline int divup(int a, int b) { return (a + b - 1) / b; }

// ---------------- elementwise convert ----------------
__global__ void k_f32_to_f16(const float* __restrict__ in, half_t* __restrict__ out, int n) {
  int i = blockIdx.x * blockDim.x + threadIdx.x;
  if (i < n) out[i] = (half_t)in[i];
}

// ---- pack weights [Nout][C,KH,KW] f32 -> [NoutPad][Kpad] f16, K = (ky,kx,c) ----
// k = (ky*KWp + kx)*Cp + c ; zero outside Nout/C/KH/KW.  (KH=KW=KWp=1 => plain fc)
__global__ void k_pack_w_nhwc(const float* __restrict__ w, half_t* __restrict__ out,
                              int Nout, int NoutPad, int C, int Cp,
                              int KH, int KW, int KWp, int Kpad) {
  int i = blockIdx.x * blockDim.x + threadIdx.x;
  int total = NoutPad * Kpad;
  if (i >= total) return;
  int n = i / Kpad, k = i - n * Kpad;
  int s = k / Cp, c = k - s * Cp;
  int ky = s / KWp, kx = s - ky * KWp;
  float v = 0.0f;
  if (n < Nout && c < C && ky < KH && kx < KW)
    v = w[((n * C + c) * KH + ky) * KW + kx];
  out[i] = (half_t)v;
}

// ---------------- chunk loader for NHWC implicit-GEMM conv ----------------
// chunk j holds K in [8j, 8j+8). CP=1 uses KWp=8 row chunks (kx clamped);
// CP in {8,16}: one aligned v8h at a single (ky,kx) position (s clamped).
template <int CP, int KH, int KW>
__device__ __forceinline__ v8h load_chunk(const half_t* __restrict__ Abase, int W, int j) {
  if (CP == 1) {
    int ky = (j < KH) ? j : (KH - 1);
    const half_t* p = Abase + (long)ky * W;
    v8h v;
#pragma unroll
    for (int e = 0; e < 8; ++e) {
      int kx = (e < KW) ? e : (KW - 1);   // clamped: weight is zero there
      v[e] = p[kx];
    }
    return v;
  } else {
    int s = (j * 8) / CP;                 // CP power of two -> shift
    int off = (j * 8) - s * CP;           // 0 or 8
    int smax = KH * KW - 1;
    s = (s <= smax) ? s : smax;           // clamped: weight is zero there
    int ky = s / KW, kx = s - (s / KW) * KW;  // compile-time KW -> magic mul
    return *(const v8h*)(Abase + ((long)ky * W + kx) * CP + off);
  }
}

// ---------------- WMMA implicit-GEMM conv, NHWC in/out ----------------
// A: [N,H,W,CP] f16 ; Wp: [NoutPad][Kpad] (K=(ky,kx,c)) ; out: [N,OH,OW,OCp]
// cols in [Nout,OCp) have zero B columns -> store exact zeros (free channel pad).
template <int CP, int KH, int KW>
__global__ void k_wmma_conv(const half_t* __restrict__ A,
                            const half_t* __restrict__ Wp,
                            const float*  __restrict__ bias,
                            half_t* __restrict__ out,
                            int numMTiles, int numNTiles,
                            int Kpad, int Nout, int OCp,
                            int H, int W, int OH, int OW) {
  int tile = blockIdx.x * (blockDim.x >> 5) + (threadIdx.x >> 5);
  if (tile >= numMTiles * numNTiles) return;   // uniform per wave
  int mTile = tile / numNTiles;
  int nTile = tile - mTile * numNTiles;

  int lane = threadIdx.x & 31;
  int g    = lane >> 4;
  int lr   = lane & 15;

  int mBase = mTile * 16;
  int col   = nTile * 16 + lr;
  int rowA  = mBase + lr;

  int sp  = OH * OW;
  int img = rowA / sp;
  int r   = rowA - img * sp;
  int oy  = r / OW, ox = r - (r / OW) * OW;
  const half_t* Abase = A + ((long)img * H * W + (long)oy * W + ox) * CP;
  const half_t* Brow  = Wp + (long)col * Kpad + g * 16;

  v8f acc = {};
  int nk = Kpad >> 5;
  for (int ks = 0; ks < nk; ++ks) {
    int j0 = 4 * ks + 2 * g;  // chunk for a[0..7]  (K = k0+16g .. +7)
    int j1 = j0 + 1;          // chunk for a[8..15] (K = k0+16g+8 .. +7)
    v8h c0 = load_chunk<CP, KH, KW>(Abase, W, j0);
    v8h c1 = load_chunk<CP, KH, KW>(Abase, W, j1);
    v16h a = __builtin_shufflevector(c0, c1, 0, 1, 2, 3, 4, 5, 6, 7,
                                     8, 9, 10, 11, 12, 13, 14, 15);
    v16h b = *(const v16h*)(Brow + ks * 32);
    acc = __builtin_amdgcn_wmma_f32_16x16x32_f16(false, a, false, b, (short)0, acc,
                                                 false, false);
  }

  float bv = (col < Nout) ? bias[col] : 0.0f;
  bool store = (col < OCp);
#pragma unroll
  for (int j = 0; j < 8; ++j) {
    int m = mBase + g * 8 + j;           // C/D layout: VGPR j -> M = j + 8*group
    int im = m / sp; int rr = m - im * sp;
    if (store) out[((long)im * sp + rr) * OCp + col] = (half_t)(acc[j] + bv);
  }
}

// ---------------- WMMA dense FC: A [M][Kpad] (zero-padded), out [M][Nout] ------
__global__ void k_wmma_fc(const half_t* __restrict__ A,
                          const half_t* __restrict__ Wp,
                          const float*  __restrict__ bias,
                          half_t* __restrict__ out,
                          int numMTiles, int numNTiles,
                          int Kpad, int Nout, int relu) {
  int tile = blockIdx.x * (blockDim.x >> 5) + (threadIdx.x >> 5);
  if (tile >= numMTiles * numNTiles) return;
  int mTile = tile / numNTiles;
  int nTile = tile - mTile * numNTiles;
  int lane = threadIdx.x & 31;
  int g = lane >> 4, lr = lane & 15;
  int mBase = mTile * 16;
  int col = nTile * 16 + lr;
  int rowA = mBase + lr;

  const half_t* Arow = A + (long)rowA * Kpad + g * 16;
  const half_t* Brow = Wp + (long)col * Kpad + g * 16;

  v8f acc = {};
  int nk = Kpad >> 5;
  for (int ks = 0; ks < nk; ++ks) {
    int k0 = ks * 32;
    v8h a0 = *(const v8h*)(Arow + k0);
    v8h a1 = *(const v8h*)(Arow + k0 + 8);
    v16h a = __builtin_shufflevector(a0, a1, 0, 1, 2, 3, 4, 5, 6, 7,
                                     8, 9, 10, 11, 12, 13, 14, 15);
    v16h b = *(const v16h*)(Brow + k0);
    acc = __builtin_amdgcn_wmma_f32_16x16x32_f16(false, a, false, b, (short)0, acc,
                                                 false, false);
  }

  bool nOK = (col < Nout);
  float bv = nOK ? bias[col] : 0.0f;
#pragma unroll
  for (int j = 0; j < 8; ++j) {
    int m = mBase + g * 8 + j;
    float v = acc[j] + bv;
    if (relu) v = fmaxf(v, 0.0f);
    if (nOK) out[(long)m * Nout + col] = (half_t)v;
  }
}

// ---------------- 2x2/2 maxpool + relu, NHWC, padded row stride ----------------
// in [N,H,W,CP] -> out [N][rowStride]; r >= OH*OW*CP zero-filled (K padding).
__global__ void k_pool_relu_nhwc(const half_t* __restrict__ in, half_t* __restrict__ out,
                                 int N, int H, int W, int CP, int OH, int OW,
                                 int rowStride) {
  int i = blockIdx.x * blockDim.x + threadIdx.x;
  int total = N * rowStride;
  if (i >= total) return;
  int n = i / rowStride;
  int r = i - n * rowStride;
  int chw = OH * OW * CP;
  if (r >= chw) { out[i] = (half_t)0.0f; return; }
  int c = r % CP; int t = r / CP;
  int ow = t % OW; int oh = t / OW;
  const half_t* p = in + (((long)n * H + 2 * oh) * W + 2 * ow) * CP + c;
  float a = (float)p[0], b = (float)p[CP], c2 = (float)p[(long)W * CP],
        d = (float)p[(long)W * CP + CP];
  float m = fmaxf(fmaxf(a, b), fmaxf(c2, d));
  out[i] = (half_t)fmaxf(m, 0.0f);
}

// ---------------- per-sample dynamic 2x2 conv + pool + relu -> NHWC [N,13,13,16] ----
__global__ void k_dynconv_pool(const half_t* __restrict__ xh,   // [N,28,28]
                               const half_t* __restrict__ ker,  // [N,40]
                               half_t* __restrict__ out, int N) {
  int i = blockIdx.x * blockDim.x + threadIdx.x;
  int total = N * 169 * 16;
  if (i >= total) return;
  int c = i & 15; int t = i >> 4;
  int px = t % 13; t /= 13;
  int py = t % 13; int n = t / 13;
  if (c >= 10) { out[i] = (half_t)0.0f; return; }
  const half_t* kp = ker + (long)n * 40 + c * 4;
  float k00 = (float)kp[0], k01 = (float)kp[1], k10 = (float)kp[2], k11 = (float)kp[3];
  const half_t* xb = xh + (long)n * 784;
  float m = -1e30f;
#pragma unroll
  for (int dy = 0; dy < 2; ++dy)
#pragma unroll
    for (int dx = 0; dx < 2; ++dx) {
      int oy = 2 * py + dy, ox = 2 * px + dx;
      const half_t* q = xb + oy * 28 + ox;
      float v = (float)q[0] * k00 + (float)q[1] * k01 +
                (float)q[28] * k10 + (float)q[29] * k11;
      m = fmaxf(m, v);
    }
  out[i] = (half_t)fmaxf(m, 0.0f);
}

// ---------------- fc2 (50->10) + log_softmax, f32 out ----------------
__global__ void k_fc2_logsoftmax(const half_t* __restrict__ h,  // [N,50]
                                 const float* __restrict__ w,   // [10,50]
                                 const float* __restrict__ b,   // [10]
                                 float* __restrict__ out, int N) {
  int n = blockIdx.x * blockDim.x + threadIdx.x;
  if (n >= N) return;
  float hv[50];
#pragma unroll
  for (int k = 0; k < 50; ++k) hv[k] = (float)h[(long)n * 50 + k];
  float z[10];
  float mx = -1e30f;
#pragma unroll
  for (int j = 0; j < 10; ++j) {
    float s = b[j];
    for (int k = 0; k < 50; ++k) s += hv[k] * w[j * 50 + k];
    z[j] = s; mx = fmaxf(mx, s);
  }
  float se = 0.0f;
#pragma unroll
  for (int j = 0; j < 10; ++j) se += expf(z[j] - mx);
  float lse = logf(se);
#pragma unroll
  for (int j = 0; j < 10; ++j) out[(long)n * 10 + j] = z[j] - mx - lse;
}

extern "C" void kernel_launch(void* const* d_in, const int* in_sizes, int n_in,
                              void* d_out, int out_size, void* d_ws, size_t ws_size,
                              hipStream_t stream) {
  const float* x        = (const float*)d_in[0];
  const float* kf_w1    = (const float*)d_in[1];
  const float* kf_b1    = (const float*)d_in[2];
  const float* kf_w2    = (const float*)d_in[3];
  const float* kf_b2    = (const float*)d_in[4];
  const float* kf_fc1_w = (const float*)d_in[5];
  const float* kf_fc1_b = (const float*)d_in[6];
  const float* kf_fc2_w = (const float*)d_in[7];
  const float* kf_fc2_b = (const float*)d_in[8];
  const float* conv2_w  = (const float*)d_in[9];
  const float* conv2_b  = (const float*)d_in[10];
  const float* fc1_w    = (const float*)d_in[11];
  const float* fc1_b    = (const float*)d_in[12];
  const float* fc2_w    = (const float*)d_in[13];
  const float* fc2_b    = (const float*)d_in[14];
  float* out = (float*)d_out;

  const int N = in_sizes[0] / 784;

  char* ws = (char*)d_ws;
  size_t o = 0;
  auto alloc = [&](size_t bytes) -> char* {
    char* p = ws + o;
    o += (bytes + 255) & ~(size_t)255;
    return p;
  };
  // big region reused over time: conv1out | conv2out | dynp+conv3out
  size_t dynpB  = ((size_t)N * 169 * 16 * 2 + 255) & ~(size_t)255;   // 44.3MB
  size_t bigB   = dynpB + (size_t)N * 81 * 24 * 2;                   // + 31.9MB
  size_t c1B    = (size_t)N * 484 * 8 * 2;                           // 63.4MB
  if (c1B > bigB) bigB = c1B;

  half_t* xh    = (half_t*)alloc((size_t)N * 784 * 2);
  char*   big   = alloc(bigB);
  half_t* pool1 = (half_t*)alloc((size_t)N * 968 * 2);   // [N,11,11,8]
  half_t* pool2 = (half_t*)alloc((size_t)N * 160 * 2);   // [N,3,3,16] + pad -> 160
  half_t* h32   = (half_t*)alloc((size_t)N * 32 * 2);
  half_t* ker40 = (half_t*)alloc((size_t)N * 40 * 2);
  half_t* pool3 = (half_t*)alloc((size_t)N * 384 * 2);   // [N,4,4,24]
  half_t* h50   = (half_t*)alloc((size_t)N * 50 * 2);
  half_t* w1p   = (half_t*)alloc(16 * 64 * 2);
  half_t* w2p   = (half_t*)alloc(16 * 224 * 2);
  half_t* w3p   = (half_t*)alloc(32 * 416 * 2);
  half_t* wf1p  = (half_t*)alloc(32 * 160 * 2);
  half_t* wf2p  = (half_t*)alloc(48 * 32 * 2);
  half_t* wg1p  = (half_t*)alloc(64 * 384 * 2);

  half_t* conv1out = (half_t*)big;                 // [N,22,22,8]   steps 3-4
  half_t* conv2out = (half_t*)big;                 // [N,7,7,16]    steps 5-6
  half_t* dynp     = (half_t*)big;                 // [N,13,13,16]  steps 9-10
  half_t* conv3out = (half_t*)(big + dynpB);       // [N,9,9,24]    steps 10-11

  const int B = 256;

  // 1) x -> f16
  {
    int n = N * 784;
    k_f32_to_f16<<<divup(n, B), B, 0, stream>>>(x, xh, n);
  }
  // 2) pack weights (K = (ky,kx,c) order)
  k_pack_w_nhwc<<<divup(16 * 64, B), B, 0, stream>>>(kf_w1,    w1p,  8, 16, 1, 1, 7, 7, 8, 64);
  k_pack_w_nhwc<<<divup(16 * 224, B), B, 0, stream>>>(kf_w2,   w2p, 10, 16, 8, 8, 5, 5, 5, 224);
  k_pack_w_nhwc<<<divup(32 * 416, B), B, 0, stream>>>(conv2_w, w3p, 20, 32, 10, 16, 5, 5, 5, 416);
  k_pack_w_nhwc<<<divup(32 * 160, B), B, 0, stream>>>(kf_fc1_w, wf1p, 32, 32, 10, 16, 3, 3, 3, 160);
  k_pack_w_nhwc<<<divup(48 * 32, B), B, 0, stream>>>(kf_fc2_w, wf2p, 40, 48, 32, 32, 1, 1, 1, 32);
  k_pack_w_nhwc<<<divup(64 * 384, B), B, 0, stream>>>(fc1_w,   wg1p, 50, 64, 20, 24, 4, 4, 4, 384);

  // 3) conv1: [N,28,28,1] *(8,1,7,7) -> [N,22,22,8]
  {
    int mt = (N * 484) / 16, nt = 1;
    k_wmma_conv<1, 7, 7><<<divup(mt * nt, 8), B, 0, stream>>>(
        xh, w1p, kf_b1, conv1out, mt, nt, 64, 8, 8, 28, 28, 22, 22);
  }
  // 4) pool+relu -> [N,11,11,8]
  k_pool_relu_nhwc<<<divup(N * 968, B), B, 0, stream>>>(conv1out, pool1,
                                                        N, 22, 22, 8, 11, 11, 968);
  // 5) conv2: [N,11,11,8] *(10,8,5,5) -> [N,7,7,16] (ch 10..15 = 0)
  {
    int mt = (N * 49) / 16, nt = 1;
    k_wmma_conv<8, 5, 5><<<divup(mt * nt, 8), B, 0, stream>>>(
        pool1, w2p, kf_b2, conv2out, mt, nt, 224, 10, 16, 11, 11, 7, 7);
  }
  // 6) pool+relu -> [N,160] = [N,3,3,16] zero-padded rows
  k_pool_relu_nhwc<<<divup(N * 160, B), B, 0, stream>>>(conv2out, pool2,
                                                        N, 7, 7, 16, 3, 3, 160);
  // 7) kf_fc1: [N,160]->[N,32] relu
  k_wmma_fc<<<divup((N / 16) * 2, 8), B, 0, stream>>>(pool2, wf1p, kf_fc1_b, h32,
                                                      N / 16, 2, 160, 32, 1);
  // 8) kf_fc2: [N,32]->[N,40] (dynamic kernels)
  k_wmma_fc<<<divup((N / 16) * 3, 8), B, 0, stream>>>(h32, wf2p, kf_fc2_b, ker40,
                                                      N / 16, 3, 32, 40, 0);
  // 9) per-sample dynamic conv + pool + relu -> [N,13,13,16]
  k_dynconv_pool<<<divup(N * 169 * 16, B), B, 0, stream>>>(xh, ker40, dynp, N);
  // 10) conv3: [N,13,13,16] *(20,10,5,5) -> [N,9,9,24] (ch 20..23 = 0)
  {
    int mt = (N * 81) / 16, nt = 2;
    k_wmma_conv<16, 5, 5><<<divup(mt * nt, 8), B, 0, stream>>>(
        dynp, w3p, conv2_b, conv3out, mt, nt, 416, 20, 24, 13, 13, 9, 9);
  }
  // 11) pool+relu -> [N,384] = [N,4,4,24]
  k_pool_relu_nhwc<<<divup(N * 384, B), B, 0, stream>>>(conv3out, pool3,
                                                        N, 9, 9, 24, 4, 4, 384);
  // 12) fc1: [N,384]->[N,50] relu
  k_wmma_fc<<<divup((N / 16) * 4, 8), B, 0, stream>>>(pool3, wg1p, fc1_b, h50,
                                                      N / 16, 4, 384, 50, 1);
  // 13) fc2 + log_softmax -> out [N,10] f32
  k_fc2_logsoftmax<<<divup(N, B), B, 0, stream>>>(h50, fc2_w, fc2_b, out, N);
}
